// LegendreKANLayer_21680994910846
// MI455X (gfx1250) — matrix operand order repaired and checked by reading, not compiled, not run on hardware
//
#include <hip/hip_runtime.h>
#include <hip/hip_bf16.h>

// Problem dims (fixed by reference)
#define BDIM   4096     // batch
#define IDIM   1024     // in features
#define ODIM   1024     // out features
#define DDIM   6        // degree+1
#define KBASE  IDIM                 // 1024 : base GEMM K
#define KLEG   (IDIM * DDIM)        // 6144 : legendre GEMM K
#define KTOT   (KBASE + KLEG)       // 7168 : fused K
#define BK     64
#define NKCH   (KTOT / BK)          // 112 K-chunks
#define KC_BASE_END (KBASE / BK - 1) // chunk 15 ends the base GEMM

typedef __attribute__((ext_vector_type(16))) __bf16 bf16x16;
typedef __attribute__((ext_vector_type(8)))  float  f32x8;

__device__ __forceinline__ unsigned short f2bf(float f) {
    unsigned int u = __float_as_uint(f);
    unsigned int r = u + 0x7FFFu + ((u >> 16) & 1u);   // round-to-nearest-even
    return (unsigned short)(r >> 16);
}

__device__ __forceinline__ float silu_fast(float c) {
    // c * rcp(1 + exp(-c)) : v_exp_f32 + v_rcp_f32 + v_mul, no IEEE div expansion
    return c * __builtin_amdgcn_rcpf(1.0f + __expf(-c));
}

// ---------------------------------------------------------------------------
// Prep A: per batch row, min/max normalize, Legendre recurrence, fold x into
// basis, emit bf16 row [ x(1024) | (x*P_d)(i,d interleaved, 6144) ].
// One block (256 threads) per batch row.
// ---------------------------------------------------------------------------
__global__ __launch_bounds__(256) void kan_prep_a(const float* __restrict__ x,
                                                  unsigned short* __restrict__ A) {
    const int b = blockIdx.x;
    const int t = threadIdx.x;
    __shared__ float red[256];
    __shared__ float s_min, s_max;

    const float* row = x + (size_t)b * IDIM;
    float vals[4];
    float vmin =  3.0e38f, vmax = -3.0e38f;
    #pragma unroll
    for (int j = 0; j < 4; ++j) {
        float v = row[t + 256 * j];
        vals[j] = v;
        vmin = fminf(vmin, v);
        vmax = fmaxf(vmax, v);
    }
    red[t] = vmin; __syncthreads();
    for (int s = 128; s > 0; s >>= 1) {
        if (t < s) red[t] = fminf(red[t], red[t + s]);
        __syncthreads();
    }
    if (t == 0) s_min = red[0];
    __syncthreads();
    red[t] = vmax; __syncthreads();
    for (int s = 128; s > 0; s >>= 1) {
        if (t < s) red[t] = fmaxf(red[t], red[t + s]);
        __syncthreads();
    }
    if (t == 0) s_max = red[0];
    __syncthreads();

    const float mn  = s_min;
    const float inv = 2.0f / (s_max - mn + 1e-7f);
    unsigned short* Arow = A + (size_t)b * KTOT;

    #pragma unroll
    for (int j = 0; j < 4; ++j) {
        const int col = t + 256 * j;
        const float v  = vals[j];
        const float xn = (v - mn) * inv - 1.0f;
        float p[DDIM];
        p[0] = 1.0f;
        p[1] = xn;
        #pragma unroll
        for (int n = 2; n <= 5; ++n)
            p[n] = ((float)(2 * n - 1) * xn * p[n - 1] - (float)(n - 1) * p[n - 2]) / (float)n;
        Arow[col] = f2bf(v);
        unsigned short* dst = Arow + KBASE + col * DDIM;
        #pragma unroll
        for (int d = 0; d < DDIM; ++d) dst[d] = f2bf(v * p[d]);
    }
}

// ---------------------------------------------------------------------------
// Prep B: build Bt[o][k] (N-major, K contiguous) in bf16:
//   k <  1024 : base_weight[k][o]
//   k >= 1024 : legendre_weight[i][o][d] with k-1024 = i*6 + d
// ---------------------------------------------------------------------------
__global__ __launch_bounds__(256) void kan_prep_b(const float* __restrict__ bw,
                                                  const float* __restrict__ lw,
                                                  unsigned short* __restrict__ Bt) {
    const int idx = blockIdx.x * 256 + threadIdx.x;   // ODIM * KTOT threads
    const int o = idx / KTOT;
    const int k = idx - o * KTOT;
    float v;
    if (k < KBASE) {
        v = bw[(size_t)k * ODIM + o];
    } else {
        const int kk = k - KBASE;
        const int i  = kk / DDIM;
        const int d  = kk - i * DDIM;
        v = lw[((size_t)i * ODIM + o) * DDIM + d];
    }
    Bt[idx] = f2bf(v);
}

// ---------------------------------------------------------------------------
// Fused bf16 WMMA GEMM over K = 7168 with silu boundary at K = 1024.
// Block: 256 threads = 8 waves; tile 128(M) x 128(N) x 64(K).
// Wave grid 2(M) x 4(N); each wave: 4x2 tiles, 16 v_wmma per K-chunk.
// At the base/legendre boundary silu(acc) is written to out; epilogue adds
// the legendre accumulator (SCALE_BASE = SCALE_LEGENDRE = 1).
// ---------------------------------------------------------------------------
__global__ __launch_bounds__(256) void kan_gemm(const unsigned short* __restrict__ A,
                                                const unsigned short* __restrict__ Bt,
                                                float* __restrict__ out) {
    __shared__ unsigned short As[128][BK];   // 16 KB
    __shared__ unsigned short Bs[128][BK];   // 16 KB

    const int t      = threadIdx.x;
    const int wave   = t >> 5;
    const int lane   = t & 31;
    const int laneLo = lane & 15;
    const int laneHi = lane >> 4;
    const int waveM  = (wave & 1) * 64;      // 2 waves along M
    const int waveN  = (wave >> 1) * 32;     // 4 waves along N
    const int bBase  = blockIdx.y * 128;
    const int oBase  = blockIdx.x * 128;

    // global->LDS load mapping: 2 threads per 128B row, 64B (4x b128) each
    const int grow = t >> 1;                 // 0..127
    const int gseg = (t & 1) * 32;           // bf16 column offset 0/32

    const size_t aGlob = (size_t)(bBase + grow) * KTOT + gseg;
    const size_t bGlob = (size_t)(oBase + grow) * KTOT + gseg;

    f32x8 acc[4][2] = {};

    union Frag { uint4 q[2]; bf16x16 v; };

    for (int kc = 0; kc < NKCH; ++kc) {
        const int k0 = kc * BK;
        __syncthreads();
        {   // A tile: 128x64 bf16
            const uint4* src = (const uint4*)(A + aGlob + k0);
            uint4 v0 = src[0], v1 = src[1], v2 = src[2], v3 = src[3];
            uint4* dst = (uint4*)(&As[grow][gseg]);
            dst[0] = v0; dst[1] = v1; dst[2] = v2; dst[3] = v3;
        }
        {   // B tile: 128x64 bf16
            const uint4* src = (const uint4*)(Bt + bGlob + k0);
            uint4 v0 = src[0], v1 = src[1], v2 = src[2], v3 = src[3];
            uint4* dst = (uint4*)(&Bs[grow][gseg]);
            dst[0] = v0; dst[1] = v1; dst[2] = v2; dst[3] = v3;
        }
        if (kc + 1 < NKCH) {   // warm caches for next chunk -> global_prefetch_b8
            __builtin_prefetch(A  + aGlob + k0 + BK, 0, 1);
            __builtin_prefetch(Bt + bGlob + k0 + BK, 0, 1);
        }
        __syncthreads();

        #pragma unroll
        for (int kk = 0; kk < 2; ++kk) {     // two 16x16x32 K-steps per chunk
            // A frags: lane laneLo = row M; laneHi picks K halves
            // {0..7,16..23} vs {8..15,24..31} within this K-step  [ISA layout]
            Frag a[4], bb[2];
            #pragma unroll
            for (int m = 0; m < 4; ++m) {
                const uint4* p = (const uint4*)(&As[waveM + m * 16 + laneLo][kk * 32 + laneHi * 8]);
                a[m].q[0] = p[0];   // K = laneHi*8 .. +7
                a[m].q[1] = p[2];   // K = laneHi*8 + 16 .. +23
            }
            // B frags: lane laneLo = N column, laneHi picks K {0..7,16..23} / {8..15,24..31}
            #pragma unroll
            for (int n = 0; n < 2; ++n) {
                const uint4* p = (const uint4*)(&Bs[waveN + n * 16 + laneLo][kk * 32 + laneHi * 8]);
                bb[n].q[0] = p[0];
                bb[n].q[1] = p[2];
            }
            #pragma unroll
            for (int m = 0; m < 4; ++m)
                #pragma unroll
                for (int n = 0; n < 2; ++n)
                    acc[m][n] = __builtin_amdgcn_wmma_f32_16x16x32_bf16(
                        false, a[m].v, false, bb[n].v, (short)0, acc[m][n], false, false);
        }

        if (kc == KC_BASE_END) {
            // Base GEMM done: write silu(acc) to out, reset accumulators.
            #pragma unroll
            for (int m = 0; m < 4; ++m)
                #pragma unroll
                for (int n = 0; n < 2; ++n) {
                    const int row = bBase + waveM + m * 16 + laneHi * 8;
                    const int col = oBase + waveN + n * 16 + laneLo;
                    #pragma unroll
                    for (int r = 0; r < 8; ++r)
                        out[(size_t)(row + r) * ODIM + col] = silu_fast(acc[m][n][r]);
                    acc[m][n] = (f32x8){};
                }
        }
    }

    // Epilogue: out += legendre accumulator.
    // C/D layout — VGPR r, lanes0-15: M=r N=lane; lanes16-31: M=8+r.
    #pragma unroll
    for (int m = 0; m < 4; ++m)
        #pragma unroll
        for (int n = 0; n < 2; ++n) {
            const int row = bBase + waveM + m * 16 + laneHi * 8;
            const int col = oBase + waveN + n * 16 + laneLo;
            #pragma unroll
            for (int r = 0; r < 8; ++r) {
                float* p = out + (size_t)(row + r) * ODIM + col;
                *p = *p + acc[m][n][r];
            }
        }
}

extern "C" void kernel_launch(void* const* d_in, const int* in_sizes, int n_in,
                              void* d_out, int out_size, void* d_ws, size_t ws_size,
                              hipStream_t stream) {
    (void)in_sizes; (void)n_in; (void)out_size; (void)ws_size;
    const float* x  = (const float*)d_in[0];                 // [4096,1024]
    const float* bw = (const float*)d_in[1];                 // [1024,1024]
    const float* lw = (const float*)d_in[2];                 // [1024,1024,6]
    float* out = (float*)d_out;                              // [4096,1024]

    unsigned short* A  = (unsigned short*)d_ws;                                    // 4096*7168 bf16 = 58.7 MB
    unsigned short* Bt = (unsigned short*)((char*)d_ws + (size_t)BDIM * KTOT * 2); // 1024*7168 bf16 = 14.7 MB

    kan_prep_a<<<BDIM, 256, 0, stream>>>(x, A);
    kan_prep_b<<<(ODIM * KTOT) / 256, 256, 0, stream>>>(bw, lw, Bt);
    kan_gemm<<<dim3(ODIM / 128, BDIM / 128), 256, 0, stream>>>(A, Bt, out);
}
